// DependencyViTBlock_15333033247145
// MI455X (gfx1250) — compile-verified, matrix operand order checked by reading, not run on hardware
//
#include <hip/hip_runtime.h>

typedef __bf16 bf16;
typedef __attribute__((ext_vector_type(16))) __bf16 v16bf;
typedef __attribute__((ext_vector_type(8)))  float   v8f;

#define DEV static __device__ __forceinline__

constexpr int Bb   = 16;
constexpr int Nt   = 577;
constexpr int Cc   = 768;
constexpr int Hh   = 12;
constexpr int Dd   = 64;
constexpr int Mtok = Bb * Nt;        // 9232 tokens
constexpr int BH   = Bb * Hh;        // 192 (b,h) pairs
constexpr int VST  = 584;            // padded V^T row stride (16B-aligned uint4 loads)
constexpr float QSCALE = 0.125f;     // 64^-0.5

// ---------- bf16 helpers (bit-exact RNE, no reliance on cvt availability) ----------
DEV bf16 f2bf(float f) {
  unsigned u = __builtin_bit_cast(unsigned, f);
  u += 0x7fffu + ((u >> 16) & 1u);
  unsigned short s = (unsigned short)(u >> 16);
  return __builtin_bit_cast(bf16, s);
}
DEV float bf2f(bf16 h) {
  unsigned short s = __builtin_bit_cast(unsigned short, h);
  unsigned u = (unsigned)s << 16;
  return __builtin_bit_cast(float, u);
}
DEV float bflo(unsigned w) { return __builtin_bit_cast(float, w << 16); }
DEV float bfhi(unsigned w) { return __builtin_bit_cast(float, w & 0xffff0000u); }

DEV float waveSum(float v) {
#pragma unroll
  for (int off = 16; off > 0; off >>= 1) v += __shfl_down(v, off, 32);
  return v;
}
DEV float geluf(float x) { return 0.5f * x * (1.f + erff(x * 0.70710678118f)); }

DEV v8f wmma_bf16(v16bf a, v16bf b, v8f c) {
  return __builtin_amdgcn_wmma_f32_16x16x32_bf16(false, a, false, b, (short)0, c, false, false);
}

// ---------------------- f32 -> bf16 weight conversion ----------------------
__global__ __launch_bounds__(256) void cvt_kernel(const float* __restrict__ src,
                                                  bf16* __restrict__ dst, int n) {
  int i = blockIdx.x * 256 + threadIdx.x;
  if (i < n) dst[i] = f2bf(src[i]);
}

// ---------------------- LayerNorm (one block per token row) ----------------------
__global__ __launch_bounds__(256) void ln_kernel(const float* __restrict__ x,
                                                 const float* __restrict__ w,
                                                 const float* __restrict__ bvec,
                                                 bf16* __restrict__ out) {
  int row = blockIdx.x;
  int tid = threadIdx.x, lane = tid & 31, wave = tid >> 5;
  __shared__ float ps[8], pq[8], stat[2];
  const float* xr = x + (size_t)row * Cc;
  float v0 = xr[tid], v1 = xr[tid + 256], v2 = xr[tid + 512];
  float ls = v0 + v1 + v2;
  float lq = v0 * v0 + v1 * v1 + v2 * v2;
  ls = waveSum(ls); lq = waveSum(lq);
  if (lane == 0) { ps[wave] = ls; pq[wave] = lq; }
  __syncthreads();
  if (tid == 0) {
    float s = 0.f, q = 0.f;
#pragma unroll
    for (int i = 0; i < 8; ++i) { s += ps[i]; q += pq[i]; }
    float mean = s * (1.f / 768.f);
    float var  = q * (1.f / 768.f) - mean * mean;
    stat[0] = mean; stat[1] = rsqrtf(var + 1e-5f);
  }
  __syncthreads();
  float mean = stat[0], inv = stat[1];
  bf16* orow = out + (size_t)row * Cc;
  orow[tid]       = f2bf((v0 - mean) * inv * w[tid]       + bvec[tid]);
  orow[tid + 256] = f2bf((v1 - mean) * inv * w[tid + 256] + bvec[tid + 256]);
  orow[tid + 512] = f2bf((v2 - mean) * inv * w[tid + 512] + bvec[tid + 512]);
}

// ---------------- head-selector softmax + message-controller gate ----------------
__global__ __launch_bounds__(256) void hs_mc_kernel(const bf16* __restrict__ xn,
                                                    const float* __restrict__ hs_w,
                                                    const float* __restrict__ mc_w1,
                                                    const float* __restrict__ mc_w2,
                                                    const float* __restrict__ m_in,
                                                    float* __restrict__ p,
                                                    float* __restrict__ mnew,
                                                    float* __restrict__ out_mnew) {
  int row = blockIdx.x;
  int tid = threadIdx.x, lane = tid & 31, wave = tid >> 5;
  __shared__ float xs[768];
  __shared__ float hpart[12][8];
  __shared__ float gpart[8];
#pragma unroll
  for (int i = 0; i < 3; ++i) {
    int c = tid + i * 256;
    xs[c] = bf2f(xn[(size_t)row * Cc + c]);
  }
  __syncthreads();
  float hl[12];
#pragma unroll
  for (int h = 0; h < 12; ++h) hl[h] = 0.f;
#pragma unroll
  for (int i = 0; i < 3; ++i) {
    int c = tid + i * 256;
    float xv = xs[c];
#pragma unroll
    for (int h = 0; h < 12; ++h) hl[h] += xv * hs_w[h * 768 + c];
  }
#pragma unroll
  for (int h = 0; h < 12; ++h) {
    float r = waveSum(hl[h]);
    if (lane == 0) hpart[h][wave] = r;
  }
  float gl = 0.f;
  for (int j = tid; j < 384; j += 256) {
    const float* wr = mc_w1 + (size_t)j * 768;
    float s = 0.f;
    for (int c = 0; c < 768; ++c) s += xs[c] * wr[c];
    gl += geluf(s) * mc_w2[j];
  }
  gl = waveSum(gl);
  if (lane == 0) gpart[wave] = gl;
  __syncthreads();
  if (tid == 0) {
    float lg[12], mx = -1e30f;
#pragma unroll
    for (int h = 0; h < 12; ++h) {
      float s = 0.f;
#pragma unroll
      for (int i = 0; i < 8; ++i) s += hpart[h][i];
      lg[h] = s * 10.f;                       // /TEMP
      mx = fmaxf(mx, lg[h]);
    }
    float se = 0.f;
#pragma unroll
    for (int h = 0; h < 12; ++h) { lg[h] = __expf(lg[h] - mx); se += lg[h]; }
    float rse = 1.f / se;
    int b = row / Nt, tok = row - b * Nt;
#pragma unroll
    for (int h = 0; h < 12; ++h) p[((size_t)(b * Hh + h)) * Nt + tok] = lg[h] * rse;
    float g = 0.f;
#pragma unroll
    for (int i = 0; i < 8; ++i) g += gpart[i];
    float gate = 1.f / (1.f + __expf(-g));
    float mn = m_in[row] * gate;
    mnew[row] = mn;
    out_mnew[row] = mn;
  }
}

// -------------- attention pass 1: per-row online max/sum + fused coef --------------
__global__ __launch_bounds__(256) void attn_stats_kernel(const bf16* __restrict__ qg,
                                                         const bf16* __restrict__ kg,
                                                         const float* __restrict__ p,
                                                         const float* __restrict__ mnew,
                                                         float* __restrict__ rmax,
                                                         float* __restrict__ coef) {
  int bh = blockIdx.x;
  int b = bh / Hh;
  const uint4* kbase = (const uint4*)(kg + (size_t)bh * Nt * Dd);  // 8 uint4 per row
  for (int mrow = threadIdx.x; mrow < Nt; mrow += 256) {
    const uint4* qp = (const uint4*)(qg + ((size_t)bh * Nt + mrow) * Dd);
    float qv[64];
#pragma unroll
    for (int c = 0; c < 8; ++c) {
      uint4 t = qp[c];
      qv[c*8+0]=bflo(t.x); qv[c*8+1]=bfhi(t.x);
      qv[c*8+2]=bflo(t.y); qv[c*8+3]=bfhi(t.y);
      qv[c*8+4]=bflo(t.z); qv[c*8+5]=bfhi(t.z);
      qv[c*8+6]=bflo(t.w); qv[c*8+7]=bfhi(t.w);
    }
    float runm = -1e30f, runs = 0.f;
    for (int n = 0; n < Nt; ++n) {
      const uint4* kr = kbase + n * 8;
      float s = 0.f;
#pragma unroll
      for (int c = 0; c < 8; ++c) {
        uint4 t = kr[c];
        s += qv[c*8+0]*bflo(t.x) + qv[c*8+1]*bfhi(t.x)
           + qv[c*8+2]*bflo(t.y) + qv[c*8+3]*bfhi(t.y)
           + qv[c*8+4]*bflo(t.z) + qv[c*8+5]*bfhi(t.z)
           + qv[c*8+6]*bflo(t.w) + qv[c*8+7]*bfhi(t.w);
      }
      float nm = fmaxf(runm, s);
      runs = runs * __expf(runm - nm) + __expf(s - nm);
      runm = nm;
    }
    rmax[(size_t)bh * Nt + mrow] = runm;
    coef[(size_t)bh * Nt + mrow] =
        p[(size_t)bh * Nt + mrow] * mnew[(size_t)b * Nt + mrow] / runs;
  }
}

// -------------- attention pass 2: WMMA  xo = (A∘w)^T V  per (b,h) --------------
__global__ __launch_bounds__(256) void attn_pv_kernel(const bf16* __restrict__ qg,
                                                      const bf16* __restrict__ kg,
                                                      const bf16* __restrict__ vtg,
                                                      const float* __restrict__ rmax,
                                                      const float* __restrict__ coef,
                                                      bf16* __restrict__ xo) {
  int bh = blockIdx.y;
  int b = bh / Hh, h = bh - b * Hh;
  int bn0 = blockIdx.x * 128;

  __shared__ alignas(32) bf16 Ks[128 * 80];       // K rows for our n-tile (pad stride 80)
  __shared__ alignas(32) bf16 Qs[32 * 80];        // Q rows for current m-chunk
  __shared__ alignas(32) bf16 Vs[64 * 48];        // V^T rows d, cols m-chunk (pad stride 48)
  __shared__ alignas(32) bf16 Ps[8 * 16 * 48];    // per-wave P staging (A-layout source)
  __shared__ float smax[32], scoef[32];

  int tid = threadIdx.x, wave = tid >> 5, lane = tid & 31;
  int fr = lane & 15, fko = (lane >> 4) * 16;
  int nw0 = wave * 16;

  const bf16* kbase = kg  + (size_t)bh * Nt * Dd;
  const bf16* qbase = qg  + (size_t)bh * Nt * Dd;
  const bf16* vbase = vtg + (size_t)bh * Dd * VST;

  // load K tile (128 rows x 64) once
  {
    int r = tid >> 1, ch = (tid & 1) * 32;
    int n = bn0 + r;
    uint4 zz = {0u, 0u, 0u, 0u};
#pragma unroll
    for (int c = 0; c < 4; ++c) {
      uint4 dv = (n < Nt) ? ((const uint4*)(kbase + (size_t)n * Dd + ch))[c] : zz;
      *(uint4*)(Ks + r * 80 + ch + c * 8) = dv;
    }
  }

  v8f acc[4] = {};
  for (int mc = 0; mc < (Nt + 31) / 32; ++mc) {
    int m0 = mc * 32;
    __syncthreads();  // protect Qs/Vs from previous iteration readers
    {
      int r = tid >> 3, cc = (tid & 7) * 8;  // 32 rows x 64 cols, one uint4/thread
      int m = m0 + r;
      uint4 zz = {0u, 0u, 0u, 0u};
      uint4 dv = (m < Nt) ? *(const uint4*)(qbase + (size_t)m * Dd + cc) : zz;
      *(uint4*)(Qs + r * 80 + cc) = dv;
      // prefetch next m-chunk of Q into caches
      if (m + 32 < Nt) __builtin_prefetch(qbase + (size_t)(m + 32) * Dd + cc, 0, 1);
    }
    {
      int r = tid >> 2, cc = (tid & 3) * 8;  // 64 rows x 32 cols, one uint4/thread
      uint4 zz = {0u, 0u, 0u, 0u};
      uint4 dv = (m0 + cc < Nt) ? *(const uint4*)(vbase + (size_t)r * VST + m0 + cc) : zz;
      *(uint4*)(Vs + r * 48 + cc) = dv;      // tail cols >= Nt neutralized by coef==0
      if (m0 + 32 + cc < Nt) __builtin_prefetch(vbase + (size_t)r * VST + m0 + 32 + cc, 0, 1);
    }
    if (tid < 32) {
      int m = m0 + tid;
      smax[tid]  = (m < Nt) ? rmax[(size_t)bh * Nt + m] : 0.f;
      scoef[tid] = (m < Nt) ? coef[(size_t)bh * Nt + m] : 0.f;
    }
    __syncthreads();

    // S^T tiles: st[n (16), m (2x16)] over d = 64 (two WMMA k-steps)
    v8f st0 = {}, st1 = {};
#pragma unroll
    for (int kk = 0; kk < 64; kk += 32) {
      v16bf afrag = *(const v16bf*)(Ks + (nw0 + fr) * 80 + kk + fko);
      v16bf b0    = *(const v16bf*)(Qs + fr * 80 + kk + fko);
      v16bf b1    = *(const v16bf*)(Qs + (16 + fr) * 80 + kk + fko);
      st0 = wmma_bf16(afrag, b0, st0);
      st1 = wmma_bf16(afrag, b1, st1);
    }

    // P = exp(st - rowmax_m) * coef_m  (per-column), stage to LDS in A-layout order
    {
      int rbase = 8 * (lane >> 4);
      float mx0 = smax[fr],      cf0 = scoef[fr];
      float mx1 = smax[16 + fr], cf1 = scoef[16 + fr];
      bf16* prow = Ps + wave * 16 * 48;
#pragma unroll
      for (int r = 0; r < 8; ++r) {
        prow[(rbase + r) * 48 + fr]      = f2bf(__expf(st0[r] - mx0) * cf0);
        prow[(rbase + r) * 48 + 16 + fr] = f2bf(__expf(st1[r] - mx1) * cf1);
      }
    }
    __syncthreads();

    // xo[n,d] += P[n,m] * V[m,d]   (B operand from V^T rows)
    v16bf pa = *(const v16bf*)(Ps + wave * 16 * 48 + fr * 48 + fko);
#pragma unroll
    for (int u = 0; u < 4; ++u) {
      v16bf vb = *(const v16bf*)(Vs + (u * 16 + fr) * 48 + fko);
      acc[u] = wmma_bf16(pa, vb, acc[u]);
    }
  }

  // store xo as bf16 [M, C] (row = b*Nt + n, col = h*64 + d)
  int rhi = 8 * (lane >> 4);
#pragma unroll
  for (int u = 0; u < 4; ++u) {
#pragma unroll
    for (int r = 0; r < 8; ++r) {
      int n = bn0 + nw0 + r + rhi;
      if (n < Nt) {
        size_t row = (size_t)(b * Nt + n);
        xo[row * Cc + h * Dd + u * 16 + fr] = f2bf(acc[u][r]);
      }
    }
  }
}

// ---------------------- generic 128x128 WMMA GEMM, BK=64, with mode epilogues ----------------------
// MODE 0: QKV   (scale q, scatter q/k [B,H,N,D], v transposed [B,H,D,VST])
// MODE 1: PROJ  (+bias +resid -> f32 x1)
// MODE 2: FC1   (+bias, exact GELU -> bf16)
// MODE 3: FC2   (+bias +resid -> f32 d_out)
template <int MODE>
__global__ __launch_bounds__(256) void gemm_kernel(const bf16* __restrict__ A,
                                                   const bf16* __restrict__ W,   // [Nout][K]
                                                   const float* __restrict__ bias,
                                                   int K,
                                                   const float* __restrict__ resid,
                                                   float* __restrict__ outF,
                                                   bf16* __restrict__ outB,
                                                   bf16* __restrict__ qb,
                                                   bf16* __restrict__ kb,
                                                   bf16* __restrict__ vt) {
  __shared__ alignas(32) bf16 As[128 * 80];   // 128 rows x 64 (pad stride 80)
  __shared__ alignas(32) bf16 Bs[128 * 80];
  int tid = threadIdx.x, wave = tid >> 5, lane = tid & 31;
  int wr = wave & 3, wc = wave >> 2;                 // wave tile: rows wr*32, cols wc*64
  int bm0 = blockIdx.x * 128, bn0 = blockIdx.y * 128;
  int fr = lane & 15, fko = (lane >> 4) * 16;
  int lrow = tid >> 1, lcol = (tid & 1) * 32;        // each thread: 32 elems (4x uint4) per tile

  const bf16* arow = A + (size_t)(bm0 + lrow) * K + lcol;
  const bf16* wrow = W + (size_t)(bn0 + lrow) * K + lcol;
  bool avalid = (bm0 + lrow) < Mtok;

  v8f acc[2][4] = {};
  for (int k0 = 0; k0 < K; k0 += 64) {
    uint4 zz = {0u, 0u, 0u, 0u};
#pragma unroll
    for (int c = 0; c < 4; ++c) {
      uint4 av = avalid ? ((const uint4*)(arow + k0))[c] : zz;
      *(uint4*)(As + lrow * 80 + lcol + c * 8) = av;
      uint4 wv = ((const uint4*)(wrow + k0))[c];
      *(uint4*)(Bs + lrow * 80 + lcol + c * 8) = wv;
    }
    // overlap next k-step's L2->L0 latency with this step's WMMAs
    if (k0 + 64 < K) {
      __builtin_prefetch(arow + k0 + 64, 0, 1);
      __builtin_prefetch(wrow + k0 + 64, 0, 1);
    }
    __syncthreads();

#pragma unroll
    for (int kk = 0; kk < 64; kk += 32) {
      v16bf af0 = *(const v16bf*)(As + (wr * 32 + fr) * 80 + kk + fko);
      v16bf af1 = *(const v16bf*)(As + (wr * 32 + 16 + fr) * 80 + kk + fko);
#pragma unroll
      for (int u = 0; u < 4; ++u) {
        v16bf bf = *(const v16bf*)(Bs + (wc * 64 + u * 16 + fr) * 80 + kk + fko);
        acc[0][u] = wmma_bf16(af0, bf, acc[0][u]);
        acc[1][u] = wmma_bf16(af1, bf, acc[1][u]);
      }
    }
    __syncthreads();
  }

  int rhi = 8 * (lane >> 4);
#pragma unroll
  for (int t = 0; t < 2; ++t) {
#pragma unroll
    for (int u = 0; u < 4; ++u) {
      int n = bn0 + wc * 64 + u * 16 + fr;
      float bv = bias[n];
#pragma unroll
      for (int r = 0; r < 8; ++r) {
        int m = bm0 + wr * 32 + t * 16 + r + rhi;
        if (m >= Mtok) continue;
        float val = acc[t][u][r] + bv;
        if constexpr (MODE == 0) {
          int seg = n / 768;
          int rem = n - seg * 768;
          int hh = rem >> 6, d = rem & 63;
          int bb = m / Nt, tok = m - bb * Nt;
          int bh = bb * Hh + hh;
          if (seg == 0)      qb[((size_t)bh * Nt + tok) * Dd + d] = f2bf(val * QSCALE);
          else if (seg == 1) kb[((size_t)bh * Nt + tok) * Dd + d] = f2bf(val);
          else               vt[((size_t)bh * Dd + d) * VST + tok] = f2bf(val);
        } else if constexpr (MODE == 1) {
          outF[(size_t)m * Cc + n] = val + resid[(size_t)m * Cc + n];
        } else if constexpr (MODE == 2) {
          outB[(size_t)m * 3072 + n] = f2bf(geluf(val));
        } else {
          outF[(size_t)m * Cc + n] = val + resid[(size_t)m * Cc + n];
        }
      }
    }
  }
}

// ---------------------------------- launcher ----------------------------------
extern "C" void kernel_launch(void* const* d_in, const int* in_sizes, int n_in,
                              void* d_out, int out_size, void* d_ws, size_t ws_size,
                              hipStream_t stream) {
  (void)in_sizes; (void)n_in; (void)out_size; (void)ws_size;
  const float* x       = (const float*)d_in[0];
  const float* m_in    = (const float*)d_in[1];
  const float* norm1_w = (const float*)d_in[2];
  const float* norm1_b = (const float*)d_in[3];
  const float* qkv_w   = (const float*)d_in[4];
  const float* qkv_b   = (const float*)d_in[5];
  const float* hs_w    = (const float*)d_in[6];
  const float* mc_w1   = (const float*)d_in[7];
  const float* mc_w2   = (const float*)d_in[8];
  const float* proj_w  = (const float*)d_in[9];
  const float* proj_b  = (const float*)d_in[10];
  const float* norm2_w = (const float*)d_in[11];
  const float* norm2_b = (const float*)d_in[12];
  const float* fc1_w   = (const float*)d_in[13];
  const float* fc1_b   = (const float*)d_in[14];
  const float* fc2_w   = (const float*)d_in[15];
  const float* fc2_b   = (const float*)d_in[16];

  char* ws = (char*)d_ws;
  size_t off = 0;
  auto alloc = [&](size_t bytes) -> void* {
    void* p = ws + off;
    off = (off + bytes + 255) & ~(size_t)255;
    return p;
  };
  bf16*  wqkv = (bf16*)alloc((size_t)3 * Cc * Cc * 2);
  bf16*  wprj = (bf16*)alloc((size_t)Cc * Cc * 2);
  bf16*  wf1  = (bf16*)alloc((size_t)4 * Cc * Cc * 2);
  bf16*  wf2  = (bf16*)alloc((size_t)4 * Cc * Cc * 2);
  bf16*  xn   = (bf16*)alloc((size_t)Mtok * Cc * 2);
  bf16*  qb   = (bf16*)alloc((size_t)BH * Nt * Dd * 2);
  bf16*  kb   = (bf16*)alloc((size_t)BH * Nt * Dd * 2);
  bf16*  vt   = (bf16*)alloc((size_t)BH * Dd * VST * 2);
  float* pbuf = (float*)alloc((size_t)BH * Nt * 4);
  float* mnew = (float*)alloc((size_t)Mtok * 4);
  float* rmax = (float*)alloc((size_t)BH * Nt * 4);
  float* coef = (float*)alloc((size_t)BH * Nt * 4);
  bf16*  xo   = (bf16*)alloc((size_t)Mtok * Cc * 2);
  float* x1   = (float*)alloc((size_t)Mtok * Cc * 4);
  bf16*  hbuf = (bf16*)alloc((size_t)Mtok * 4 * Cc * 2);

  float* out_x  = (float*)d_out;
  float* out_mn = out_x + (size_t)Mtok * Cc;

  // 1) weight conversion f32 -> bf16
  {
    int n;
    n = 3 * Cc * Cc; cvt_kernel<<<(n + 255) / 256, 256, 0, stream>>>(qkv_w,  wqkv, n);
    n = Cc * Cc;     cvt_kernel<<<(n + 255) / 256, 256, 0, stream>>>(proj_w, wprj, n);
    n = 4 * Cc * Cc; cvt_kernel<<<(n + 255) / 256, 256, 0, stream>>>(fc1_w,  wf1,  n);
    n = 4 * Cc * Cc; cvt_kernel<<<(n + 255) / 256, 256, 0, stream>>>(fc2_w,  wf2,  n);
  }

  const int MT = (Mtok + 127) / 128;  // 73

  // 2) LN1
  ln_kernel<<<Mtok, 256, 0, stream>>>(x, norm1_w, norm1_b, xn);
  // 3) QKV GEMM (WMMA)
  gemm_kernel<0><<<dim3(MT, (3 * Cc) / 128), 256, 0, stream>>>(
      xn, wqkv, qkv_b, Cc, nullptr, nullptr, nullptr, qb, kb, vt);
  // 4) head selector + message controller
  hs_mc_kernel<<<Mtok, 256, 0, stream>>>(xn, hs_w, mc_w1, mc_w2, m_in, pbuf, mnew, out_mn);
  // 5) attention pass 1: row stats + fused coefficients
  attn_stats_kernel<<<BH, 256, 0, stream>>>(qb, kb, pbuf, mnew, rmax, coef);
  // 6) attention pass 2: WMMA PV
  attn_pv_kernel<<<dim3((Nt + 127) / 128, BH), 256, 0, stream>>>(qb, kb, vt, rmax, coef, xo);
  // 7) proj GEMM + residual -> x1
  gemm_kernel<1><<<dim3(MT, Cc / 128), 256, 0, stream>>>(
      xo, wprj, proj_b, Cc, x, x1, nullptr, nullptr, nullptr, nullptr);
  // 8) LN2
  ln_kernel<<<Mtok, 256, 0, stream>>>(x1, norm2_w, norm2_b, xn);
  // 9) fc1 GEMM + GELU
  gemm_kernel<2><<<dim3(MT, (4 * Cc) / 128), 256, 0, stream>>>(
      xn, wf1, fc1_b, Cc, nullptr, nullptr, hbuf, nullptr, nullptr, nullptr);
  // 10) fc2 GEMM + residual -> output
  gemm_kernel<3><<<dim3(MT, Cc / 128), 256, 0, stream>>>(
      hbuf, wf2, fc2_b, 4 * Cc, x1, out_x, nullptr, nullptr, nullptr, nullptr);
}
